// GNS_80109730005535
// MI455X (gfx1250) — matrix operand order repaired and checked by reading, not compiled, not run on hardware
//
#include <hip/hip_runtime.h>

typedef _Float16 v16h __attribute__((ext_vector_type(16)));
typedef _Float16 v8h  __attribute__((ext_vector_type(8)));
typedef float    v8f  __attribute__((ext_vector_type(8)));

#define WAIT_DS() asm volatile("s_wait_dscnt 0" ::: "memory")

static constexpr int N_NODES = 100000;
static constexpr int N_EDGES = 400000;
static constexpr int LAT     = 128;   // latent width
static constexpr int NT      = 8;     // 8 n-tiles of 16 cols = 128
static constexpr int WPB     = 4;     // waves per block (128 threads)

__device__ __forceinline__ float eluf(float x) { return x > 0.f ? x : __expf(x) - 1.f; }

__device__ __forceinline__ void atomic_add_f32(float* p, float v) {
  unsafeAtomicAdd(p, v);   // hardware global_atomic_add_f32 on CDNA5
}

// ---- A fragment (16x32 f16): lane<16 -> K {kb+0..7, kb+16..23}; lane>=16 -> K {kb+8..15, kb+24..31}
__device__ __forceinline__ v16h load_a_h16(const _Float16* __restrict__ row, int kbase, int up) {
  int off = kbase + (up ? 8 : 0);
  v8h lo = *(const v8h*)(row + off);
  v8h hi = *(const v8h*)(row + off + 16);
  return __builtin_shufflevector(lo, hi, 0,1,2,3,4,5,6,7,8,9,10,11,12,13,14,15);
}

__device__ __forceinline__ v16h load_a_f32(const float* __restrict__ row, int kbase, int up) {
  int off = kbase + (up ? 8 : 0);
  v16h a;
#pragma unroll
  for (int i = 0; i < 8; ++i) { a[i] = (_Float16)row[off + i]; a[i + 8] = (_Float16)row[off + 16 + i]; }
  return a;
}

__device__ __forceinline__ void init_acc(v8f* acc, const float* __restrict__ b, int lr) {
#pragma unroll
  for (int nt = 0; nt < NT; ++nt) {
    float bv = b[nt * 16 + lr];
    v8f t;
#pragma unroll
    for (int r = 0; r < 8; ++r) t[r] = bv;
    acc[nt] = t;
  }
}

// ---- one K-step (32) for a 32-row M-tile: one B fragment feeds two WMMAs
__device__ __forceinline__ void gemm_step2(v16h a0, v16h a1, const _Float16* __restrict__ pw,
                                           int ks, int lane, v8f* acc0, v8f* acc1) {
#pragma unroll
  for (int nt = 0; nt < NT; ++nt) {
    v16h b = *(const v16h*)(pw + ((size_t)(ks * NT + nt) * 32 + lane) * 16);
    acc0[nt] = __builtin_amdgcn_wmma_f32_16x16x32_f16(false, a0, false, b, (short)0, acc0[nt], false, false);
    acc1[nt] = __builtin_amdgcn_wmma_f32_16x16x32_f16(false, a1, false, b, (short)0, acc1[nt], false, false);
  }
}

// ---- D-layout -> ELU -> LDS row-major at row offset
__device__ __forceinline__ void tile_elu_to_lds(_Float16* __restrict__ my, const v8f* acc,
                                                int lr, int up, int rowoff) {
#pragma unroll
  for (int nt = 0; nt < NT; ++nt)
#pragma unroll
    for (int r = 0; r < 8; ++r)
      my[(rowoff + (up ? r + 8 : r)) * LAT + nt * 16 + lr] = (_Float16)eluf(acc[nt][r]);
}

// =============== weight pre-pack into B-fragment layout ===============
// B frag (32x16 f16): lane<16 -> K {kb+0..7, kb+16..23}, lane>=16 -> K {kb+8..15, kb+24..31}; N = nt*16 + (lane&15)
__global__ void k_pack_w(const float* __restrict__ W, int din, int dout,
                         int ksteps, int ntiles, _Float16* __restrict__ dst) {
  int t = blockIdx.x * blockDim.x + threadIdx.x;
  int total = ksteps * ntiles * 512;
  if (t >= total) return;
  int j = t & 15;
  int l = (t >> 4) & 31;
  int f = t >> 9;                 // ks*ntiles + nt
  int nt = f % ntiles;
  int ks = f / ntiles;
  int r = ks * 32 + ((l < 16) ? 0 : 8) + ((j < 8) ? j : j + 8);
  int c = nt * 16 + (l & 15);
  float v = (r < din && c < dout) ? W[(size_t)r * dout + c] : 0.f;
  dst[((size_t)(f) * 32 + l) * 16 + j] = (_Float16)v;
}

__global__ void k_zero_f32(float* __restrict__ p, int n) {
  int i = blockIdx.x * blockDim.x + threadIdx.x;
  if (i < n) p[i] = 0.f;
}

// =============== encoder: h = elu( elu(x@W1+b1) @ W2 + b2 ), Din=5, 32 rows/wave ===============
__global__ void __launch_bounds__(128) k_encoder(
    const float* __restrict__ x,
    const _Float16* __restrict__ pW1, const float* __restrict__ b1,
    const _Float16* __restrict__ pW2, const float* __restrict__ b2,
    _Float16* __restrict__ h16, int n_tiles) {
  __shared__ __align__(32) _Float16 stg[WPB][32 * LAT];
  int lane = threadIdx.x & 31, wid = threadIdx.x >> 5;
  int tile = blockIdx.x * WPB + wid;
  if (tile >= n_tiles) return;
  int up = lane >> 4, lr = lane & 15;
  _Float16* my = stg[wid];
  int base = tile * 32;

  v16h a0, a1;
#pragma unroll
  for (int i = 0; i < 16; ++i) { a0[i] = (_Float16)0.f; a1[i] = (_Float16)0.f; }
  if (!up) {  // K 0..4 valid, only low-half lanes carry K<8
    const float* x0 = x + (size_t)(base + lr) * 5;
    const float* x1 = x + (size_t)(base + 16 + lr) * 5;
#pragma unroll
    for (int i = 0; i < 5; ++i) { a0[i] = (_Float16)x0[i]; a1[i] = (_Float16)x1[i]; }
  }
  v8f acc0[NT], acc1[NT];
  init_acc(acc0, b1, lr);
  init_acc(acc1, b1, lr);
  gemm_step2(a0, a1, pW1, 0, lane, acc0, acc1);
  tile_elu_to_lds(my, acc0, lr, up, 0);
  tile_elu_to_lds(my, acc1, lr, up, 16);
  WAIT_DS();

  init_acc(acc0, b2, lr);
  init_acc(acc1, b2, lr);
#pragma unroll
  for (int ks = 0; ks < 4; ++ks)
    gemm_step2(load_a_h16(my + lr * LAT, ks * 32, up),
               load_a_h16(my + (16 + lr) * LAT, ks * 32, up), pW2, ks, lane, acc0, acc1);

#pragma unroll
  for (int nt = 0; nt < NT; ++nt)
#pragma unroll
    for (int r = 0; r < 8; ++r) {
      int rr = (up ? r + 8 : r);
      h16[(size_t)(base + rr) * LAT + nt * 16 + lr]      = (_Float16)eluf(acc0[nt][r]);
      h16[(size_t)(base + 16 + rr) * LAT + nt * 16 + lr] = (_Float16)eluf(acc1[nt][r]);
    }
}

// =============== edge message MLP (259->128->128->128) + scatter-add, 32 edges/wave ===============
__global__ void __launch_bounds__(128) k_message(
    const _Float16* __restrict__ h16, const float* __restrict__ ea,
    const int* __restrict__ eidx,
    const _Float16* __restrict__ pW1, const float* __restrict__ b1,
    const _Float16* __restrict__ pW2, const float* __restrict__ b2,
    const _Float16* __restrict__ pW3, const float* __restrict__ b3,
    float* __restrict__ agg, int n_tiles) {
  __shared__ __align__(32) _Float16 stg[WPB][32 * LAT];
  int lane = threadIdx.x & 31, wid = threadIdx.x >> 5;
  int tile = blockIdx.x * WPB + wid;
  if (tile >= n_tiles) return;
  int up = lane >> 4, lr = lane & 15;
  _Float16* my = stg[wid];
  int e0 = tile * 32 + lr;
  int e1 = e0 + 16;
  int s0 = eidx[e0],           s1 = eidx[e1];
  int d0 = eidx[N_EDGES + e0], d1 = eidx[N_EDGES + e1];
  const _Float16* hs0 = h16 + (size_t)s0 * LAT;
  const _Float16* hs1 = h16 + (size_t)s1 * LAT;
  const _Float16* hd0 = h16 + (size_t)d0 * LAT;
  const _Float16* hd1 = h16 + (size_t)d1 * LAT;

  v8f acc0[NT], acc1[NT];
  init_acc(acc0, b1, lr);
  init_acc(acc1, b1, lr);
#pragma unroll
  for (int ks = 0; ks < 4; ++ks)        // h[src]
    gemm_step2(load_a_h16(hs0, ks * 32, up), load_a_h16(hs1, ks * 32, up),
               pW1, ks, lane, acc0, acc1);
#pragma unroll
  for (int ks = 0; ks < 4; ++ks)        // h[dst]
    gemm_step2(load_a_h16(hd0, ks * 32, up), load_a_h16(hd1, ks * 32, up),
               pW1, ks + 4, lane, acc0, acc1);
  {                                     // edge_attr (K 256..258)
    v16h a0, a1;
#pragma unroll
    for (int i = 0; i < 16; ++i) { a0[i] = (_Float16)0.f; a1[i] = (_Float16)0.f; }
    if (!up) {
      const float* er0 = ea + (size_t)e0 * 3;
      const float* er1 = ea + (size_t)e1 * 3;
#pragma unroll
      for (int i = 0; i < 3; ++i) { a0[i] = (_Float16)er0[i]; a1[i] = (_Float16)er1[i]; }
    }
    gemm_step2(a0, a1, pW1, 8, lane, acc0, acc1);
  }
  tile_elu_to_lds(my, acc0, lr, up, 0);
  tile_elu_to_lds(my, acc1, lr, up, 16);
  WAIT_DS();

  init_acc(acc0, b2, lr);
  init_acc(acc1, b2, lr);
#pragma unroll
  for (int ks = 0; ks < 4; ++ks)
    gemm_step2(load_a_h16(my + lr * LAT, ks * 32, up),
               load_a_h16(my + (16 + lr) * LAT, ks * 32, up), pW2, ks, lane, acc0, acc1);
  tile_elu_to_lds(my, acc0, lr, up, 0);   // LDS pipe in-order: reads above retire first
  tile_elu_to_lds(my, acc1, lr, up, 16);
  WAIT_DS();

  init_acc(acc0, b3, lr);
  init_acc(acc1, b3, lr);
#pragma unroll
  for (int ks = 0; ks < 4; ++ks)
    gemm_step2(load_a_h16(my + lr * LAT, ks * 32, up),
               load_a_h16(my + (16 + lr) * LAT, ks * 32, up), pW3, ks, lane, acc0, acc1);

  // scatter-add messages into agg[dst]
#pragma unroll
  for (int r = 0; r < 8; ++r) {
    int rr  = up ? r + 8 : r;
    int dr0 = __shfl(d0, rr, 16);
    int dr1 = __shfl(d1, rr, 16);
#pragma unroll
    for (int nt = 0; nt < NT; ++nt) {
      atomic_add_f32(&agg[(size_t)dr0 * LAT + nt * 16 + lr], acc0[nt][r]);
      atomic_add_f32(&agg[(size_t)dr1 * LAT + nt * 16 + lr], acc1[nt][r]);
    }
  }
}

// =============== node update MLP (256->128->128->128) + ELU + GroupNorm, 32 nodes/wave ===============
__global__ void __launch_bounds__(128) k_update(
    _Float16* __restrict__ h16, const float* __restrict__ agg,
    const _Float16* __restrict__ pW1, const float* __restrict__ b1,
    const _Float16* __restrict__ pW2, const float* __restrict__ b2,
    const _Float16* __restrict__ pW3, const float* __restrict__ b3,
    const float* __restrict__ gamma, const float* __restrict__ beta,
    int n_tiles) {
  __shared__ __align__(32) _Float16 stg[WPB][32 * LAT];
  int lane = threadIdx.x & 31, wid = threadIdx.x >> 5;
  int tile = blockIdx.x * WPB + wid;
  if (tile >= n_tiles) return;
  int up = lane >> 4, lr = lane & 15;
  _Float16* my = stg[wid];
  int base = tile * 32;
  const _Float16* h0 = h16 + (size_t)(base + lr) * LAT;
  const _Float16* h1 = h16 + (size_t)(base + 16 + lr) * LAT;
  const float*    a0 = agg + (size_t)(base + lr) * LAT;
  const float*    a1 = agg + (size_t)(base + 16 + lr) * LAT;

  v8f acc0[NT], acc1[NT];
  init_acc(acc0, b1, lr);
  init_acc(acc1, b1, lr);
#pragma unroll
  for (int ks = 0; ks < 4; ++ks)        // h
    gemm_step2(load_a_h16(h0, ks * 32, up), load_a_h16(h1, ks * 32, up),
               pW1, ks, lane, acc0, acc1);
#pragma unroll
  for (int ks = 0; ks < 4; ++ks)        // agg (f32 -> f16)
    gemm_step2(load_a_f32(a0, ks * 32, up), load_a_f32(a1, ks * 32, up),
               pW1, ks + 4, lane, acc0, acc1);
  tile_elu_to_lds(my, acc0, lr, up, 0);
  tile_elu_to_lds(my, acc1, lr, up, 16);
  WAIT_DS();

  init_acc(acc0, b2, lr);
  init_acc(acc1, b2, lr);
#pragma unroll
  for (int ks = 0; ks < 4; ++ks)
    gemm_step2(load_a_h16(my + lr * LAT, ks * 32, up),
               load_a_h16(my + (16 + lr) * LAT, ks * 32, up), pW2, ks, lane, acc0, acc1);
  tile_elu_to_lds(my, acc0, lr, up, 0);
  tile_elu_to_lds(my, acc1, lr, up, 16);
  WAIT_DS();

  init_acc(acc0, b3, lr);
  init_acc(acc1, b3, lr);
#pragma unroll
  for (int ks = 0; ks < 4; ++ks)
    gemm_step2(load_a_h16(my + lr * LAT, ks * 32, up),
               load_a_h16(my + (16 + lr) * LAT, ks * 32, up), pW3, ks, lane, acc0, acc1);

  // outer ELU + GroupNorm (group g == n-tile g: 16 channels == 16 lanes of one half)
#pragma unroll
  for (int nt = 0; nt < NT; ++nt) {
    float g  = gamma[nt * 16 + lr];
    float bb = beta[nt * 16 + lr];
#pragma unroll
    for (int r = 0; r < 8; ++r) {
      int rr = up ? r + 8 : r;
      {
        float v = eluf(acc0[nt][r]);
        float s1 = v, s2 = v * v;
#pragma unroll
        for (int m = 1; m < 16; m <<= 1) { s1 += __shfl_xor(s1, m, 16); s2 += __shfl_xor(s2, m, 16); }
        float mu  = s1 * 0.0625f;
        float var = s2 * 0.0625f - mu * mu;
        float y = (v - mu) * rsqrtf(var + 1e-5f) * g + bb;
        h16[(size_t)(base + rr) * LAT + nt * 16 + lr] = (_Float16)y;
      }
      {
        float v = eluf(acc1[nt][r]);
        float s1 = v, s2 = v * v;
#pragma unroll
        for (int m = 1; m < 16; m <<= 1) { s1 += __shfl_xor(s1, m, 16); s2 += __shfl_xor(s2, m, 16); }
        float mu  = s1 * 0.0625f;
        float var = s2 * 0.0625f - mu * mu;
        float y = (v - mu) * rsqrtf(var + 1e-5f) * g + bb;
        h16[(size_t)(base + 16 + rr) * LAT + nt * 16 + lr] = (_Float16)y;
      }
    }
  }
}

// =============== decoder: elu(h@W1+b1) @ W2 + b2, out 4 cols, 32 nodes/wave ===============
__global__ void __launch_bounds__(128) k_decoder(
    const _Float16* __restrict__ h16,
    const _Float16* __restrict__ pW1, const float* __restrict__ b1,
    const _Float16* __restrict__ pW2, const float* __restrict__ b2,
    float* __restrict__ out, int n_tiles) {
  __shared__ __align__(32) _Float16 stg[WPB][32 * LAT];
  int lane = threadIdx.x & 31, wid = threadIdx.x >> 5;
  int tile = blockIdx.x * WPB + wid;
  if (tile >= n_tiles) return;
  int up = lane >> 4, lr = lane & 15;
  _Float16* my = stg[wid];
  int base = tile * 32;
  const _Float16* h0 = h16 + (size_t)(base + lr) * LAT;
  const _Float16* h1 = h16 + (size_t)(base + 16 + lr) * LAT;

  v8f acc0[NT], acc1[NT];
  init_acc(acc0, b1, lr);
  init_acc(acc1, b1, lr);
#pragma unroll
  for (int ks = 0; ks < 4; ++ks)
    gemm_step2(load_a_h16(h0, ks * 32, up), load_a_h16(h1, ks * 32, up),
               pW1, ks, lane, acc0, acc1);
  tile_elu_to_lds(my, acc0, lr, up, 0);
  tile_elu_to_lds(my, acc1, lr, up, 16);
  WAIT_DS();

  v8f o0, o1;
  {
    float bv = (lr < 4) ? b2[lr] : 0.f;
#pragma unroll
    for (int r = 0; r < 8; ++r) { o0[r] = bv; o1[r] = bv; }
  }
#pragma unroll
  for (int ks = 0; ks < 4; ++ks) {
    v16h t0 = load_a_h16(my + lr * LAT, ks * 32, up);
    v16h t1 = load_a_h16(my + (16 + lr) * LAT, ks * 32, up);
    v16h b = *(const v16h*)(pW2 + ((size_t)ks * 32 + lane) * 16);   // single n-tile
    o0 = __builtin_amdgcn_wmma_f32_16x16x32_f16(false, t0, false, b, (short)0, o0, false, false);
    o1 = __builtin_amdgcn_wmma_f32_16x16x32_f16(false, t1, false, b, (short)0, o1, false, false);
  }
  if (lr < 4) {
#pragma unroll
    for (int r = 0; r < 8; ++r) {
      int rr = up ? r + 8 : r;
      out[(size_t)(base + rr) * 4 + lr]      = o0[r];
      out[(size_t)(base + 16 + rr) * 4 + lr] = o1[r];
    }
  }
}

extern "C" void kernel_launch(void* const* d_in, const int* in_sizes, int n_in,
                              void* d_out, int out_size, void* d_ws, size_t ws_size,
                              hipStream_t stream) {
  (void)in_sizes; (void)n_in; (void)out_size; (void)ws_size;
  const float* x     = (const float*)d_in[0];
  const float* ea    = (const float*)d_in[1];
  const float* encW0 = (const float*)d_in[2];
  const float* encB0 = (const float*)d_in[3];
  const float* encW1 = (const float*)d_in[4];
  const float* encB1 = (const float*)d_in[5];
  const float *msgW[4][3], *msgB[4][3], *updW[4][3], *updB[4][3];
  for (int l = 0; l < 4; ++l)
    for (int k = 0; k < 3; ++k) {
      msgW[l][k] = (const float*)d_in[6 + l * 6 + k * 2];
      msgB[l][k] = (const float*)d_in[7 + l * 6 + k * 2];
      updW[l][k] = (const float*)d_in[30 + l * 6 + k * 2];
      updB[l][k] = (const float*)d_in[31 + l * 6 + k * 2];
    }
  const float* decW0 = (const float*)d_in[54];
  const float* decB0 = (const float*)d_in[55];
  const float* decW1 = (const float*)d_in[56];
  const float* decB1 = (const float*)d_in[57];
  const float* gamma = (const float*)d_in[58];
  const float* beta  = (const float*)d_in[59];
  const int*   eidx  = (const int*)d_in[60];
  float* out = (float*)d_out;

  char* wsp = (char*)d_ws;
  auto alloc = [&](size_t bytes) -> void* {
    void* p = wsp;
    wsp += (bytes + 255) & ~(size_t)255;
    return p;
  };
  auto packsz = [](int ks, int nt) -> size_t { return (size_t)ks * nt * 512 * 2; };

  _Float16* h16 = (_Float16*)alloc((size_t)N_NODES * LAT * 2);
  float*    agg = (float*)alloc((size_t)N_NODES * LAT * 4);
  _Float16* pEnc1 = (_Float16*)alloc(packsz(1, 8));
  _Float16* pEnc2 = (_Float16*)alloc(packsz(4, 8));
  _Float16 *pMsg[4][3], *pUpd[4][3];
  for (int l = 0; l < 4; ++l) {
    pMsg[l][0] = (_Float16*)alloc(packsz(9, 8));
    pMsg[l][1] = (_Float16*)alloc(packsz(4, 8));
    pMsg[l][2] = (_Float16*)alloc(packsz(4, 8));
    pUpd[l][0] = (_Float16*)alloc(packsz(8, 8));
    pUpd[l][1] = (_Float16*)alloc(packsz(4, 8));
    pUpd[l][2] = (_Float16*)alloc(packsz(4, 8));
  }
  _Float16* pDec1 = (_Float16*)alloc(packsz(4, 8));
  _Float16* pDec2 = (_Float16*)alloc(packsz(4, 1));

  auto pack = [&](const float* W, int din, int dout, int ks, int nt, _Float16* dst) {
    int total = ks * nt * 512;
    k_pack_w<<<(total + 255) / 256, 256, 0, stream>>>(W, din, dout, ks, nt, dst);
  };
  pack(encW0, 5, 128, 1, 8, pEnc1);
  pack(encW1, 128, 128, 4, 8, pEnc2);
  for (int l = 0; l < 4; ++l) {
    pack(msgW[l][0], 259, 128, 9, 8, pMsg[l][0]);
    pack(msgW[l][1], 128, 128, 4, 8, pMsg[l][1]);
    pack(msgW[l][2], 128, 128, 4, 8, pMsg[l][2]);
    pack(updW[l][0], 256, 128, 8, 8, pUpd[l][0]);
    pack(updW[l][1], 128, 128, 4, 8, pUpd[l][1]);
    pack(updW[l][2], 128, 128, 4, 8, pUpd[l][2]);
  }
  pack(decW0, 128, 128, 4, 8, pDec1);
  pack(decW1, 128, 4, 4, 1, pDec2);

  const int node_tiles = N_NODES / 32;   // 3125
  const int edge_tiles = N_EDGES / 32;   // 12500
  const int nblk = (node_tiles + WPB - 1) / WPB;
  const int eblk = (edge_tiles + WPB - 1) / WPB;

  k_encoder<<<nblk, 128, 0, stream>>>(x, pEnc1, encB0, pEnc2, encB1, h16, node_tiles);
  for (int l = 0; l < 4; ++l) {
    k_zero_f32<<<(N_NODES * LAT + 255) / 256, 256, 0, stream>>>(agg, N_NODES * LAT);
    k_message<<<eblk, 128, 0, stream>>>(h16, ea, eidx,
                                        pMsg[l][0], msgB[l][0],
                                        pMsg[l][1], msgB[l][1],
                                        pMsg[l][2], msgB[l][2],
                                        agg, edge_tiles);
    k_update<<<nblk, 128, 0, stream>>>(h16, agg,
                                       pUpd[l][0], updB[l][0],
                                       pUpd[l][1], updB[l][1],
                                       pUpd[l][2], updB[l][2],
                                       gamma, beta, node_tiles);
  }
  k_decoder<<<nblk, 128, 0, stream>>>(h16, pDec1, decB0, pDec2, decB1, out, node_tiles);
}
